// FieldFormerMLPSparse_67748814127611
// MI455X (gfx1250) — matrix operand order repaired and checked by reading, not compiled, not run on hardware
//
#include <hip/hip_runtime.h>

// ---------------- types ----------------
typedef __attribute__((ext_vector_type(16))) _Float16 v16h;
typedef __attribute__((ext_vector_type(2)))  _Float16 h2;
typedef __attribute__((ext_vector_type(8)))  float    v8f;
typedef __attribute__((ext_vector_type(4)))  unsigned int v4u;

struct __align__(16) Frag32 { v4u lo, hi; };

__device__ __forceinline__ v16h load_frag(const _Float16* p0, const _Float16* p1) {
  Frag32 f;
  f.lo = *(const v4u*)p0;
  f.hi = *(const v4u*)p1;
  return __builtin_bit_cast(v16h, f);
}

// Fast GELU: tanh-form reduced to x*sigmoid(2u), hardware exp2 + rcp.
// gelu(x) ~= x * (1 - 1/(1 + e^{2u})), u = 0.79788456*(x + 0.044715 x^3)
__device__ __forceinline__ float gelu_fast(float x) {
  float s = x * x;
  float u = x * fmaf(s, 0.0356774081f, 0.7978845608f);
  float e = __builtin_amdgcn_exp2f(u * 2.8853900817779268f);  // e^{2u}
  float r = __builtin_amdgcn_rcpf(1.0f + e);
  return fmaf(-x, r, x);  // x * (1 - r)
}

// problem constants
#define QK    64      // neighbors per query
#define DDIM  128
#define DFF   512
#define NBLK  3

// LDS layout (bytes)
#define TOK_OFF   0u        // 64*4 floats
#define POOL_OFF  1024u     // 256 floats
#define PH_OFF    2048u     // 256 floats
#define RED_OFF   3072u     // 256 floats
#define SCAL_OFF  4096u     // 2 floats (+pad)
#define HF_OFF    4112u     // 64*132 floats (stride 132, bank-rotating)
#define HN_OFF    37904u    // 64*136 halves (stride 136, 16B aligned rows)
#define T1_OFF    55312u    // 64*520 halves (stride 520)
#define SMEM_BYTES 121872u

// ---------------- weight pre-pass: f32 -> f16 (transposed / packed) ----------------
__global__ void convert_weights(const float* __restrict__ Wa,
                                const float* __restrict__ Wb,
                                const float* __restrict__ Wh1,
                                _Float16* __restrict__ WaT,
                                _Float16* __restrict__ WbT,
                                _Float16* __restrict__ Wh1P) {
  int e = blockIdx.x * blockDim.x + threadIdx.x;
  if (e < NBLK * DDIM * DFF) {
    // WaT[i][n][k] = Wa[i][k][n]   (n<512, k<128)
    int i = e / (DDIM * DFF);
    int r = e % (DDIM * DFF);
    int n = r / DDIM;
    int k = r % DDIM;
    WaT[e] = (_Float16)Wa[i * (DDIM * DFF) + k * DFF + n];
    // WbT[i][n][k] = Wb[i][k][n]   (n<128, k<512)
    int n2 = r / DFF;
    int k2 = r % DFF;
    WbT[e] = (_Float16)Wb[i * (DDIM * DFF) + k2 * DDIM + n2];
  }
  if (e < 256 * 256) {
    // Wh1P pair e=(j,n): (Wh1[j][n], Wh1[j][n+256]), j<256, n<256
    int j = e >> 8;
    int n = e & 255;
    Wh1P[2 * e + 0] = (_Float16)Wh1[j * DFF + n];
    Wh1P[2 * e + 1] = (_Float16)Wh1[j * DFF + n + 256];
  }
}

// ---------------- main fused kernel: one workgroup per query ----------------
__global__ void __launch_bounds__(256)
fieldformer_main(const float* __restrict__ obs_coords,
                 const float* __restrict__ obs_vals,
                 const float* __restrict__ log_gammas,
                 const float* __restrict__ W_in,
                 const float* __restrict__ b_in,
                 const float* __restrict__ ln_w,
                 const float* __restrict__ ln_b,
                 const float* __restrict__ ba,
                 const float* __restrict__ bb,
                 const float* __restrict__ lnh_w,
                 const float* __restrict__ lnh_b,
                 const float* __restrict__ bh1,
                 const float* __restrict__ Wh2,
                 const float* __restrict__ bh2,
                 const int*   __restrict__ q_lin,
                 const int*   __restrict__ nb_idx,
                 const _Float16* __restrict__ WaT,
                 const _Float16* __restrict__ WbT,
                 const _Float16* __restrict__ Wh1P,
                 float* __restrict__ out) {
  extern __shared__ char smem[];
  float*    tok  = (float*)(smem + TOK_OFF);
  float*    pool = (float*)(smem + POOL_OFF);
  float*    ph   = (float*)(smem + PH_OFF);
  float*    red  = (float*)(smem + RED_OFF);
  float*    scal = (float*)(smem + SCAL_OFF);
  float*    hF   = (float*)(smem + HF_OFF);     // [64][132]
  _Float16* hn   = (_Float16*)(smem + HN_OFF);  // [64][136]
  _Float16* t1h  = (_Float16*)(smem + T1_OFF);  // [64][520]

  const int q    = blockIdx.x;
  const int tid  = threadIdx.x;
  const int lane = tid & 31;
  const int wv   = tid >> 5;
  const int lrow = lane & 15;
  const int lhi  = lane >> 4;

  // ---- Stage A: gather + periodic-wrap relative features ----
  if (tid < QK) {
    int qi = q_lin[q];
    float qx = obs_coords[qi * 3 + 0];
    float qy = obs_coords[qi * 3 + 1];
    float qt = obs_coords[qi * 3 + 2];
    int nb = nb_idx[q * QK + tid];
    float nx = obs_coords[nb * 3 + 0];
    float ny = obs_coords[nb * 3 + 1];
    float nt = obs_coords[nb * 3 + 2];
    float vv = obs_vals[nb];
    float g0 = expf(log_gammas[0]);
    float g1 = expf(log_gammas[1]);
    float g2 = expf(log_gammas[2]);
    float ux = nx - qx + 0.5f; ux -= floorf(ux);
    float uy = ny - qy + 0.5f; uy -= floorf(uy);
    tok[tid * 4 + 0] = (ux - 0.5f) * g0;
    tok[tid * 4 + 1] = (uy - 0.5f) * g1;
    tok[tid * 4 + 2] = (nt - qt) * g2;
    tok[tid * 4 + 3] = vv;
  }
  __syncthreads();

  // ---- Stage A2: h = gelu(tokens @ W_in + b_in), f32 into LDS ----
  {
    int row = tid >> 2;
    int cb  = (tid & 3) * 32;
    float t0 = tok[row * 4 + 0];
    float t1 = tok[row * 4 + 1];
    float t2 = tok[row * 4 + 2];
    float t3 = tok[row * 4 + 3];
#pragma unroll 4
    for (int j = 0; j < 32; ++j) {
      int c = cb + j;
      float a = b_in[c] + t0 * W_in[c] + t1 * W_in[DDIM + c]
                        + t2 * W_in[2 * DDIM + c] + t3 * W_in[3 * DDIM + c];
      hF[row * 132 + c] = gelu_fast(a);
    }
  }
  __syncthreads();

  // ---- Stage B: 3 blocks of LN -> GEMM1 -> gelu -> GEMM2 -> gelu ----
  for (int i = 0; i < NBLK; ++i) {
    // B1: LayerNorm, fully parallel (4 lanes per row, shfl-xor butterfly)
    {
      int row = tid >> 2;
      int cb  = (tid & 3) * 32;
      float xs[32];
      float s = 0.f, s2 = 0.f;
#pragma unroll
      for (int j = 0; j < 32; ++j) {
        float x = hF[row * 132 + cb + j];
        xs[j] = x;
        s += x; s2 += x * x;
      }
      s  += __shfl_xor(s, 1);  s  += __shfl_xor(s, 2);
      s2 += __shfl_xor(s2, 1); s2 += __shfl_xor(s2, 2);
      float m = s * (1.0f / DDIM);
      float v = s2 * (1.0f / DDIM) - m * m;
      float r = rsqrtf(v + 1e-5f);
#pragma unroll
      for (int j = 0; j < 32; ++j) {
        int c = cb + j;
        hn[row * 136 + c] =
            (_Float16)(((xs[j] - m) * r) * ln_w[i * DDIM + c] + ln_b[i * DDIM + c]);
      }
    }
    __syncthreads();

    // B2: GEMM1  [64,128] x [128,512] -> gelu -> t1 (f16)
    // Wave w owns N-tiles {w, w+8, w+16, w+24}; process them in pairs so one
    // A-fragment LDS load feeds two WMMAs.
#pragma unroll
    for (int tp = 0; tp < 2; ++tp) {
      int tn0 = wv + tp * 16;  // pair (tn0, tn0+8)
      int n0 = tn0 * 16 + lrow;
      int n1 = (tn0 + 8) * 16 + lrow;
      float ba0 = ba[i * DFF + n0];
      float ba1 = ba[i * DFF + n1];
      const _Float16* B0 = WaT + (size_t)i * (DDIM * DFF) + (size_t)n0 * DDIM;
      const _Float16* B1 = WaT + (size_t)i * (DDIM * DFF) + (size_t)n1 * DDIM;
      v8f acc0[4], acc1[4];
#pragma unroll
      for (int tm = 0; tm < 4; ++tm) {
        acc0[tm] = (v8f){0.f, 0.f, 0.f, 0.f, 0.f, 0.f, 0.f, 0.f};
        acc1[tm] = (v8f){0.f, 0.f, 0.f, 0.f, 0.f, 0.f, 0.f, 0.f};
      }
#pragma unroll
      for (int kb = 0; kb < DDIM; kb += 32) {
        v16h bf0 = load_frag(B0 + kb + lhi * 16, B0 + kb + lhi * 16 + 8);
        v16h bf1 = load_frag(B1 + kb + lhi * 16, B1 + kb + lhi * 16 + 8);
#pragma unroll
        for (int tm = 0; tm < 4; ++tm) {
          const _Float16* Ar = hn + (tm * 16 + lrow) * 136;
          v16h af = load_frag(Ar + kb + lhi * 16, Ar + kb + lhi * 16 + 8);
          acc0[tm] = __builtin_amdgcn_wmma_f32_16x16x32_f16(
              false, af, false, bf0, (short)0, acc0[tm], false, false);
          acc1[tm] = __builtin_amdgcn_wmma_f32_16x16x32_f16(
              false, af, false, bf1, (short)0, acc1[tm], false, false);
        }
      }
#pragma unroll
      for (int tm = 0; tm < 4; ++tm)
#pragma unroll
        for (int v = 0; v < 8; ++v) {
          int m = tm * 16 + lhi * 8 + v;
          t1h[m * 520 + n0] = (_Float16)gelu_fast(acc0[tm][v] + ba0);
          t1h[m * 520 + n1] = (_Float16)gelu_fast(acc1[tm][v] + ba1);
        }
    }
    __syncthreads();

    // B3: GEMM2  [64,512] x [512,128] -> gelu -> hF (f32)
    {
      int n = wv * 16 + lrow;  // wave w owns N-tile w (8 waves, 128 cols)
      float bbv = bb[i * DDIM + n];
      const _Float16* Bbase = WbT + (size_t)i * (DDIM * DFF) + (size_t)n * DFF;
      v8f acc[4];
#pragma unroll
      for (int tm = 0; tm < 4; ++tm)
        acc[tm] = (v8f){0.f, 0.f, 0.f, 0.f, 0.f, 0.f, 0.f, 0.f};
      for (int kb = 0; kb < DFF; kb += 32) {
        v16h bf = load_frag(Bbase + kb + lhi * 16, Bbase + kb + lhi * 16 + 8);
#pragma unroll
        for (int tm = 0; tm < 4; ++tm) {
          const _Float16* Ar = t1h + (tm * 16 + lrow) * 520;
          v16h af = load_frag(Ar + kb + lhi * 16, Ar + kb + lhi * 16 + 8);
          acc[tm] = __builtin_amdgcn_wmma_f32_16x16x32_f16(
              false, af, false, bf, (short)0, acc[tm], false, false);
        }
      }
#pragma unroll
      for (int tm = 0; tm < 4; ++tm)
#pragma unroll
        for (int v = 0; v < 8; ++v) {
          int m = tm * 16 + lhi * 8 + v;
          hF[m * 132 + n] = gelu_fast(acc[tm][v] + bbv);
        }
    }
    __syncthreads();
  }

  // ---- Stage C: mean + max pool over the 64 neighbors ----
  if (tid < DDIM) {
    float s = 0.f;
    for (int r = 0; r < QK; ++r) s += hF[r * 132 + tid];
    pool[tid] = s * (1.0f / QK);
  } else {
    int c = tid - DDIM;
    float mx = -3.402823466e38f;
    for (int r = 0; r < QK; ++r) mx = fmaxf(mx, hF[r * 132 + c]);
    pool[DDIM + c] = mx;
  }
  __syncthreads();

  // ---- Stage D: LN over 256 pooled features ----
  if (tid < 32) {
    float s = 0.f, s2 = 0.f;
#pragma unroll
    for (int j = 0; j < 8; ++j) {
      float x = pool[tid + 32 * j];
      s += x; s2 += x * x;
    }
    red[tid] = s;
    red[32 + tid] = s2;
  }
  __syncthreads();
  if (tid == 0) {
    float s = 0.f, s2 = 0.f;
    for (int j = 0; j < 32; ++j) { s += red[j]; s2 += red[32 + j]; }
    float m = s * (1.0f / 256.0f);
    float v = s2 * (1.0f / 256.0f) - m * m;
    scal[0] = m;
    scal[1] = rsqrtf(v + 1e-5f);
  }
  __syncthreads();
  {
    float m = scal[0], r = scal[1];
    ph[tid] = (pool[tid] - m) * r * lnh_w[tid] + lnh_b[tid];
  }
  __syncthreads();

  // ---- Stage E: head MLP (M=1, VALU, packed-f16 weights) + reduction ----
  {
    const h2* Wp = (const h2*)Wh1P;  // [256][256] pairs
    float a0 = bh1[tid];
    float a1 = bh1[tid + 256];
    for (int j = 0; j < 256; ++j) {
      float p = ph[j];
      h2 w = Wp[j * 256 + tid];
      a0 = fmaf(p, (float)w.x, a0);
      a1 = fmaf(p, (float)w.y, a1);
    }
    red[tid] = gelu_fast(a0) * Wh2[tid] + gelu_fast(a1) * Wh2[tid + 256];
  }
  __syncthreads();
  for (int s = 128; s > 0; s >>= 1) {
    if (tid < s) red[tid] += red[tid + s];
    __syncthreads();
  }
  if (tid == 0) out[q] = red[0] + bh2[0];
}

// ---------------- launch ----------------
extern "C" void kernel_launch(void* const* d_in, const int* in_sizes, int n_in,
                              void* d_out, int out_size, void* d_ws, size_t ws_size,
                              hipStream_t stream) {
  const float* obs_coords = (const float*)d_in[0];
  const float* obs_vals   = (const float*)d_in[1];
  const float* log_gammas = (const float*)d_in[2];
  const float* W_in       = (const float*)d_in[3];
  const float* b_in       = (const float*)d_in[4];
  const float* ln_w       = (const float*)d_in[5];
  const float* ln_b       = (const float*)d_in[6];
  const float* Wa         = (const float*)d_in[7];
  const float* ba         = (const float*)d_in[8];
  const float* Wb         = (const float*)d_in[9];
  const float* bb         = (const float*)d_in[10];
  const float* lnh_w      = (const float*)d_in[11];
  const float* lnh_b      = (const float*)d_in[12];
  const float* Wh1        = (const float*)d_in[13];
  const float* bh1        = (const float*)d_in[14];
  const float* Wh2        = (const float*)d_in[15];
  const float* bh2        = (const float*)d_in[16];
  const int*   q_lin      = (const int*)d_in[17];
  const int*   nb_idx     = (const int*)d_in[18];
  float* out = (float*)d_out;

  const int Q = in_sizes[17];

  _Float16* WaT  = (_Float16*)d_ws;
  _Float16* WbT  = WaT + (size_t)NBLK * DDIM * DFF;
  _Float16* Wh1P = WbT + (size_t)NBLK * DDIM * DFF;

  const int convN = NBLK * DDIM * DFF;
  convert_weights<<<(convN + 255) / 256, 256, 0, stream>>>(Wa, Wb, Wh1, WaT, WbT, Wh1P);

  fieldformer_main<<<Q, 256, SMEM_BYTES, stream>>>(
      obs_coords, obs_vals, log_gammas, W_in, b_in, ln_w, ln_b,
      ba, bb, lnh_w, lnh_b, bh1, Wh2, bh2,
      q_lin, nb_idx, WaT, WbT, Wh1P, out);
}
